// MyMutilHeadAttention_13589276525293
// MI455X (gfx1250) — matrix operand order repaired and checked by reading, not compiled, not run on hardware
//
#include <hip/hip_runtime.h>

#define BB 4
#define SQ 2048
#define SKV 2048
#define DM 1024
#define NH 16
#define HD 64

typedef __attribute__((ext_vector_type(16))) _Float16 v16h;
typedef __attribute__((ext_vector_type(8)))  _Float16 v8h;
typedef __attribute__((ext_vector_type(8)))  float    v8f;

union AF { v16h v; v8h h[2]; };

#define WMMA(a, b, c) __builtin_amdgcn_wmma_f32_16x16x32_f16( \
    false, (a), false, (b), (short)0, (c), false, false)

static __device__ __forceinline__ v8h load8(const _Float16* p) {
    return *(const v8h*)p;
}

// A fragment: 16x32 f16 tile, element (m, k) at base[m*ld + k].
// Lane L holds row m = L&15; lanes 0-15 carry K=0..7 & 16..23,
// lanes 16-31 carry K=8..15 & 24..31 (CDNA5 16-bit A layout).
static __device__ __forceinline__ v16h loadA(const _Float16* base, int ld, int lane) {
    int m = lane & 15, hi = lane >> 4;
    AF a;
    a.h[0] = load8(base + (size_t)m * ld + 8 * hi);
    a.h[1] = load8(base + (size_t)m * ld + 16 + 8 * hi);
    return a.v;
}

// B fragment: 32x16 f16 tile, element (k, n) at base[n*ld + k]
// (each B column contiguous). Lanes 0-15 carry K=0..7 & 16..23, lanes 16-31 K=8..15 & 24..31.
static __device__ __forceinline__ v16h loadB(const _Float16* base, int ld, int lane) {
    int n = lane & 15, hi = lane >> 4;
    AF b;
    b.h[0] = load8(base + (size_t)n * ld + 8 * hi);
    b.h[1] = load8(base + (size_t)n * ld + 16 + 8 * hi);
    return b.v;
}

// Async global -> LDS copy of one 16B chunk (ASYNCcnt-tracked, no VGPR data).
static __device__ __forceinline__ void async_b128(const _Float16* g, _Float16* l) {
    unsigned lofs = (unsigned)(size_t)l;  // low 32 bits of generic addr = LDS offset
    asm volatile("global_load_async_to_lds_b128 %0, %1, off"
                 :: "v"(lofs), "v"((unsigned long long)(size_t)g) : "memory");
}

// ---------------- Phase 0: conversion / layout ----------------

// 8 elements per thread: float4 x2 in, v8h (b128) out.
__global__ __launch_bounds__(256) void k_f32_to_f16(const float* __restrict__ s,
                                                    _Float16* __restrict__ d, int n8) {
    int i = blockIdx.x * 256 + threadIdx.x;
    if (i >= n8) return;
    const float4* sp = (const float4*)(s + (size_t)i * 8);
    float4 f0 = sp[0], f1 = sp[1];
    v8h o;
    o[0] = (_Float16)f0.x; o[1] = (_Float16)f0.y; o[2] = (_Float16)f0.z; o[3] = (_Float16)f0.w;
    o[4] = (_Float16)f1.x; o[5] = (_Float16)f1.y; o[6] = (_Float16)f1.z; o[7] = (_Float16)f1.w;
    *(v8h*)(d + (size_t)i * 8) = o;
}

// LDS-tiled transpose: vt[((b*NH+h)*HD + d)*SKV + k] = V[b, k, h*HD + d]
// Tile: 64 k-rows x 64 d-cols per (b,h). Coalesced reads and 32B-contig writes.
__global__ __launch_bounds__(256) void k_transpose_v(const float* __restrict__ v,
                                                     _Float16* __restrict__ vt) {
    __shared__ _Float16 lds[64][72];  // pad to dodge bank conflicts
    const int t = threadIdx.x;
    const int kt = blockIdx.x % (SKV / 64);
    const int bh = blockIdx.x / (SKV / 64);
    const int b = bh / NH, h = bh % NH;
    const int k0 = kt * 64;
    const float* src = v + (size_t)b * SKV * DM + h * HD;

    for (int i = 0; i < 4; ++i) {
        int idx4 = t + i * 256;            // 1024 float4 chunks
        int kl = idx4 >> 4;                // 0..63
        int dq = (idx4 & 15) * 4;          // 0..60
        float4 f = *(const float4*)(src + (size_t)(k0 + kl) * DM + dq);
        lds[kl][dq + 0] = (_Float16)f.x; lds[kl][dq + 1] = (_Float16)f.y;
        lds[kl][dq + 2] = (_Float16)f.z; lds[kl][dq + 3] = (_Float16)f.w;
    }
    __syncthreads();
    int dl = t >> 2;                        // 0..63
    int kg = (t & 3) * 16;                  // 0,16,32,48
    _Float16* dst = vt + ((size_t)bh * HD + dl) * SKV + k0 + kg;
    v8h o0, o1;
    for (int j = 0; j < 8; ++j) { o0[j] = lds[kg + j][dl]; o1[j] = lds[kg + 8 + j][dl]; }
    *(v8h*)dst = o0;
    *(v8h*)(dst + 8) = o1;
}

// ---------------- Pass 1: per-column (query-axis) softmax stats ----------------
// One wave per 64 kv-columns: K fragments pinned in registers (8 x v16h),
// Q tiles streamed once per 64 columns.

__global__ __launch_bounds__(32) void k_colstats(const _Float16* __restrict__ qh,
                                                 const _Float16* __restrict__ kh,
                                                 float* __restrict__ colmax,
                                                 float* __restrict__ coldeninv) {
    const int lane = threadIdx.x;
    const int n = lane & 15;
    const int kt = blockIdx.x % (SKV / 64);
    const int bh = blockIdx.x / (SKV / 64);
    const int b = bh / NH, h = bh % NH;
    const int k0 = kt * 64;

    const _Float16* qb = qh + (size_t)b * SQ * DM + h * HD;
    const _Float16* kb = kh + (size_t)b * SKV * DM + h * HD;

    v16h bK[8];
    for (int c = 0; c < 4; ++c) {
        bK[2 * c]     = loadB(kb + (size_t)(k0 + 16 * c) * DM, DM, lane);       // d 0..31
        bK[2 * c + 1] = loadB(kb + (size_t)(k0 + 16 * c) * DM + 32, DM, lane);  // d 32..63
    }

    float mrun[4], srun[4];
    for (int c = 0; c < 4; ++c) { mrun[c] = -3.0e38f; srun[c] = 0.0f; }

    for (int qt = 0; qt < SQ / 16; ++qt) {
        const _Float16* qrow = qb + (size_t)(qt * 16) * DM;
        v16h a0 = loadA(qrow, DM, lane);
        v16h a1 = loadA(qrow + 32, DM, lane);
        for (int c = 0; c < 4; ++c) {
            v8f acc;
            for (int r = 0; r < 8; ++r) acc[r] = 0.0f;
            acc = WMMA(a0, bK[2 * c], acc);
            acc = WMMA(a1, bK[2 * c + 1], acc);

            float sc[8];
            float tmax = -3.0e38f;
            for (int r = 0; r < 8; ++r) {
                sc[r] = acc[r] * 0.03125f;  // 1/sqrt(1024)
                tmax = fmaxf(tmax, sc[r]);
            }
            tmax = fmaxf(tmax, __shfl_xor(tmax, 16));
            float mnew = fmaxf(mrun[c], tmax);
            float ts = 0.0f;
            for (int r = 0; r < 8; ++r) ts += __expf(sc[r] - mnew);
            ts += __shfl_xor(ts, 16);
            srun[c] = srun[c] * __expf(mrun[c] - mnew) + ts;
            mrun[c] = mnew;
        }
    }
    if (lane < 16) {
        for (int c = 0; c < 4; ++c) {
            colmax[(size_t)bh * SKV + k0 + 16 * c + n]    = mrun[c];
            coldeninv[(size_t)bh * SKV + k0 + 16 * c + n] = 1.0f / srun[c];
        }
    }
}

// ---------------- Pass 2: recompute scores, normalize, P @ V ----------------
// Workgroup = 2 waves, 64 q rows. K/V 32-k tiles double-buffered in LDS via
// async global->LDS loads: tile i+1 streams in while tile i is WMMA'd.

__global__ __launch_bounds__(64) void k_attn(const _Float16* __restrict__ qh,
                                             const _Float16* __restrict__ kh,
                                             const _Float16* __restrict__ vt,
                                             const float* __restrict__ colmax,
                                             const float* __restrict__ coldeninv,
                                             _Float16* __restrict__ ctxh) {
    __shared__ __align__(16) _Float16 ldsK[2][32 * 64];   // [k][d]
    __shared__ __align__(16) _Float16 ldsV[2][64 * 32];   // [d][k]
    __shared__ __align__(16) _Float16 ldsP[2][32 * 32];   // per-wave P [q][k]

    const int tid = threadIdx.x;
    const int wave = tid >> 5;
    const int lane = tid & 31;
    const int n = lane & 15, hi = lane >> 4;
    const int qt = blockIdx.x % (SQ / 64);
    const int bh = blockIdx.x / (SQ / 64);
    const int b = bh / NH, h = bh % NH;
    const int q0 = qt * 64 + wave * 32;  // this wave's 32 rows

    const _Float16* qb = qh + (size_t)b * SQ * DM + h * HD;
    const _Float16* kb = kh + (size_t)b * SKV * DM + h * HD;
    const _Float16* vb = vt + (size_t)bh * HD * SKV;
    const float* cm = colmax + (size_t)bh * SKV;
    const float* cd = coldeninv + (size_t)bh * SKV;

    // async-stage one 32-k K/V tile into LDS buffer `buf` (8 x b128 per thread)
    auto stage = [&](int k0, int buf) {
        for (int i = 0; i < 4; ++i) {
            int c = tid + i * 64;                 // 256 chunks: K tile 32x64 halves
            int kr = c >> 3, off = (c & 7) * 8;
            async_b128(kb + (size_t)(k0 + kr) * DM + off, &ldsK[buf][kr * 64 + off]);
        }
        for (int i = 0; i < 4; ++i) {
            int c = tid + i * 64;                 // 256 chunks: V tile 64x32 halves
            int dr = c >> 2, off = (c & 3) * 8;
            async_b128(vb + (size_t)dr * SKV + k0 + off, &ldsV[buf][dr * 32 + off]);
        }
    };

    v16h aQ[2][2];
    for (int s = 0; s < 2; ++s) {
        aQ[s][0] = loadA(qb + (size_t)(q0 + 16 * s) * DM, DM, lane);
        aQ[s][1] = loadA(qb + (size_t)(q0 + 16 * s) * DM + 32, DM, lane);
    }

    v8f acc[2][4];
    for (int s = 0; s < 2; ++s)
        for (int t = 0; t < 4; ++t)
            for (int r = 0; r < 8; ++r) acc[s][t][r] = 0.0f;

    stage(0, 0);

    for (int k0 = 0; k0 < SKV; k0 += 32) {
        const int buf = (k0 >> 5) & 1;
        asm volatile("s_wait_asynccnt 0x0" ::: "memory");  // own tile-i loads landed
        __syncthreads();  // all waves: tile i visible, tile i-1 readers done
        if (k0 + 32 < SKV) stage(k0 + 32, buf ^ 1);        // prefetch next tile

        float mA = cm[k0 + n],      dA = cd[k0 + n];
        float mB = cm[k0 + 16 + n], dB = cd[k0 + 16 + n];

        const _Float16* tK = ldsK[buf];
        const _Float16* tV = ldsV[buf];

        for (int s = 0; s < 2; ++s) {
            v8f s0, s1;
            for (int r = 0; r < 8; ++r) { s0[r] = 0.0f; s1[r] = 0.0f; }
            s0 = WMMA(aQ[s][0], loadB(tK, 64, lane), s0);
            s0 = WMMA(aQ[s][1], loadB(tK + 32, 64, lane), s0);
            s1 = WMMA(aQ[s][0], loadB(tK + 16 * 64, 64, lane), s1);
            s1 = WMMA(aQ[s][1], loadB(tK + 16 * 64 + 32, 64, lane), s1);
            for (int r = 0; r < 8; ++r) {
                ldsP[wave][(16 * s + 8 * hi + r) * 32 + n] =
                    (_Float16)(__expf(s0[r] * 0.03125f - mA) * dA);
                ldsP[wave][(16 * s + 8 * hi + r) * 32 + 16 + n] =
                    (_Float16)(__expf(s1[r] * 0.03125f - mB) * dB);
            }
        }
        asm volatile("s_wait_dscnt 0x0" ::: "memory");  // own-wave P writes visible

        for (int s = 0; s < 2; ++s) {
            const _Float16* pbase = &ldsP[wave][16 * s * 32];
            AF aP;
            aP.h[0] = *(const v8h*)&pbase[n * 32 + 8 * hi];
            aP.h[1] = *(const v8h*)&pbase[n * 32 + 16 + 8 * hi];
            for (int t = 0; t < 4; ++t)
                acc[s][t] = WMMA(aP.v, loadB(tV + t * 16 * 32, 32, lane), acc[s][t]);
        }
    }

    for (int s = 0; s < 2; ++s) {
        _Float16* cb = ctxh + ((size_t)(b * SQ + q0 + 16 * s)) * DM + h * HD;
        for (int t = 0; t < 4; ++t)
            for (int r = 0; r < 8; ++r)
                cb[(size_t)(8 * hi + r) * DM + t * 16 + n] = (_Float16)acc[s][t][r];
    }
}

// ---------------- Pass 3: out = ctx @ W^T + bias (f32 out) ----------------
// One wave per 32x64 output tile: W fragments shared across both row-subtiles.

__global__ __launch_bounds__(32) void k_linear(const _Float16* __restrict__ ctxh,
                                               const _Float16* __restrict__ wh,
                                               const float* __restrict__ bias,
                                               float* __restrict__ out) {
    const int lane = threadIdx.x;
    const int n = lane & 15, hi = lane >> 4;
    const int jt = blockIdx.x % (DM / 64);
    const int st = blockIdx.x / (DM / 64);
    const int s0 = st * 32;
    const int j0 = jt * 64;

    v8f acc[2][4];
    for (int t = 0; t < 4; ++t) {
        float bv = bias[j0 + t * 16 + n];  // bias depends only on output column
        for (int s = 0; s < 2; ++s)
            for (int r = 0; r < 8; ++r) acc[s][t][r] = bv;
    }

    for (int kk = 0; kk < DM; kk += 32) {
        v16h a0 = loadA(ctxh + (size_t)s0 * DM + kk, DM, lane);
        v16h a1 = loadA(ctxh + (size_t)(s0 + 16) * DM + kk, DM, lane);
        for (int t = 0; t < 4; ++t) {
            // B[i, j] = W[j, i]: column j contiguous in i (row-major W)
            v16h bw = loadB(wh + (size_t)(j0 + t * 16) * DM + kk, DM, lane);
            acc[0][t] = WMMA(a0, bw, acc[0][t]);
            acc[1][t] = WMMA(a1, bw, acc[1][t]);
        }
    }

    for (int s = 0; s < 2; ++s)
        for (int t = 0; t < 4; ++t)
            for (int r = 0; r < 8; ++r)
                out[(size_t)(s0 + 16 * s + 8 * hi + r) * DM + j0 + t * 16 + n] = acc[s][t][r];
}

// ---------------- launch ----------------

extern "C" void kernel_launch(void* const* d_in, const int* in_sizes, int n_in,
                              void* d_out, int out_size, void* d_ws, size_t ws_size,
                              hipStream_t stream) {
    (void)in_sizes; (void)n_in; (void)out_size; (void)ws_size;
    const float* q    = (const float*)d_in[0];
    const float* kmat = (const float*)d_in[1];
    const float* vmat = (const float*)d_in[2];
    const float* W    = (const float*)d_in[3];
    const float* bias = (const float*)d_in[4];
    float* out = (float*)d_out;

    const size_t nQKV = (size_t)BB * SQ * DM;  // 8,388,608
    _Float16* qh   = (_Float16*)d_ws;
    _Float16* kh   = qh + nQKV;
    _Float16* vt   = kh + nQKV;
    _Float16* ctxh = vt + nQKV;
    _Float16* wh   = ctxh + nQKV;
    float* colmax    = (float*)(wh + (size_t)DM * DM);
    float* coldeninv = colmax + (size_t)BB * NH * SKV;

    const int n8q = (int)(nQKV / 8);
    k_f32_to_f16<<<(n8q + 255) / 256, 256, 0, stream>>>(q, qh, n8q);
    k_f32_to_f16<<<(n8q + 255) / 256, 256, 0, stream>>>(kmat, kh, n8q);
    k_transpose_v<<<BB * NH * (SKV / 64), 256, 0, stream>>>(vmat, vt);
    const int n8w = DM * DM / 8;
    k_f32_to_f16<<<(n8w + 255) / 256, 256, 0, stream>>>(W, wh, n8w);

    k_colstats<<<BB * NH * (SKV / 64), 32, 0, stream>>>(qh, kh, colmax, coldeninv);
    k_attn<<<BB * NH * (SQ / 64), 64, 0, stream>>>(qh, kh, vt, colmax, coldeninv, ctxh);
    k_linear<<<(BB * SQ / 32) * (DM / 64), 32, 0, stream>>>(ctxh, wh, bias, out);
}
